// second_roi_27668179321337
// MI455X (gfx1250) — matrix operand order repaired and checked by reading, not compiled, not run on hardware
//
#include <hip/hip_runtime.h>

typedef _Float16 v16h __attribute__((ext_vector_type(16)));
typedef _Float16 v8h  __attribute__((ext_vector_type(8)));
typedef _Float16 v4h  __attribute__((ext_vector_type(4)));
typedef float    v8f  __attribute__((ext_vector_type(8)));

#define C_CH   512
#define HW     196
#define SP     49            // valid RoI positions
#define SPPAD  64            // padded to 4 WMMA N-tiles
#define KTILES 16            // 512 / 32
#define SMEM_BYTES 266240    // 64K bufA + 200704B Xfull (bufB/bufC alias Xfull region)
#define WT_DWORDS_PER_W 131072   // 512 tiles * 32 lanes * 8 dwords

union V16U { v16h v; v8h h[2]; };

__device__ __forceinline__ unsigned int pack2h(float a, float b) {
    union { _Float16 h[2]; unsigned int u; } u;
    u.h[0] = (_Float16)a; u.h[1] = (_Float16)b;
    return u.u;
}

// ---------------------------------------------------------------------------
// Pre-swizzle w1/w2/w3 (fp32 [512,512], row-major [out][in]) into the CDNA5
// WMMA 16-bit A-matrix lane layout, f16, one 1KB block per 16x32 tile:
//   dword index = (wsel*16384 + (mt*16+kt)*32 + lane)*8 + j
//   lane<16: K pairs {0,2,4,6,16,18,20,22}; lane>=16: +8   (ISA 7.12.2)
// ---------------------------------------------------------------------------
__global__ void __launch_bounds__(256) preswz_kernel(
    const float* __restrict__ w1, const float* __restrict__ w2,
    const float* __restrict__ w3, unsigned int* __restrict__ dst)
{
    int gid = blockIdx.x * 256 + threadIdx.x;       // 0 .. 49151
    if (gid >= 3 * 512 * 32) return;
    int wsel = gid >> 14;                           // 0..2
    int rem  = gid & 16383;
    int t    = rem >> 5;                            // tile 0..511  (t = mt*16 + kt)
    int lane = rem & 31;
    int mt = t >> 4, kt = t & 15;
    const float* src = (wsel == 0) ? w1 : ((wsel == 1) ? w2 : w3);
    int row   = mt * 16 + (lane & 15);
    int kbase = kt * 32 + ((lane & 16) ? 8 : 0);
    unsigned int* out = dst + (size_t)wsel * WT_DWORDS_PER_W + ((size_t)t * 32 + lane) * 8;
#pragma unroll
    for (int j = 0; j < 8; ++j) {
        int k = kbase + ((j < 4) ? (2 * j) : (16 + 2 * (j - 4)));
        const float* p = src + (size_t)row * 512 + k;
        out[j] = pack2h(p[0], p[1]);
    }
}

// ---------------------------------------------------------------------------
// One GEMM stage in LDS:  out[s][c] = sum_k W[c][k] * actIn[s][k] (+ bias[c])
// M=512 (out ch), N=64 (padded spatial), K=512. 8 waves, each wave owns
// 4 contiguous M-tiles x 4 N-tiles. f32 accumulate via v_wmma_f32_16x16x32_f16.
// ---------------------------------------------------------------------------
template <bool F32OUT>
__device__ __forceinline__ void gemm_stage(
    const _Float16* __restrict__ actIn,      // LDS [64][512] f16
    const unsigned int* __restrict__ wtile,  // preswizzled A tiles (global/L2)
    const float* __restrict__ bias,          // [512] fp32 (global)
    _Float16* __restrict__ outH,             // LDS [64][512] f16   (if !F32OUT)
    float* __restrict__ outF,                // LDS [64][512] f32   (if F32OUT)
    int tid)
{
    const int lane  = tid & 31;
    const int wave  = tid >> 5;
    const int dcol  = lane & 15;              // D: N within tile
    const int drow0 = (lane >> 4) << 3;       // D: first M row (0 or 8)
    const int koff  = (lane >> 4) << 3;       // B: lane<16 -> K 0..7,16..23, lane>=16 -> 8..15,24..31

    v8f acc[4][4];
#pragma unroll
    for (int mi = 0; mi < 4; ++mi)
#pragma unroll
        for (int nt = 0; nt < 4; ++nt)
            acc[mi][nt] = (v8f){0.f, 0.f, 0.f, 0.f, 0.f, 0.f, 0.f, 0.f};

    for (int kt = 0; kt < KTILES; ++kt) {
        V16U B[4];
#pragma unroll
        for (int nt = 0; nt < 4; ++nt) {
            const _Float16* pb = actIn + (nt * 16 + dcol) * C_CH + kt * 32 + koff;
            B[nt].h[0] = *(const v8h*)(pb);
            B[nt].h[1] = *(const v8h*)(pb + 16);
        }
#pragma unroll
        for (int mi = 0; mi < 4; ++mi) {
            int mt = (wave << 2) + mi;
            const unsigned int* pa = wtile + (((size_t)(mt * 16 + kt)) * 32 + lane) * 8;
            V16U A;
            A.h[0] = *(const v8h*)(pa);
            A.h[1] = *(const v8h*)(pa + 4);
#pragma unroll
            for (int nt = 0; nt < 4; ++nt)
                acc[mi][nt] = __builtin_amdgcn_wmma_f32_16x16x32_f16(
                    false, A.v, false, B[nt].v, (short)0, acc[mi][nt], false, false);
        }
    }

#pragma unroll
    for (int mi = 0; mi < 4; ++mi) {
        int c0 = ((wave << 2) + mi) * 16 + drow0;   // first out-channel this lane holds
        float bb[8];
#pragma unroll
        for (int r = 0; r < 8; ++r) bb[r] = bias[c0 + r];
#pragma unroll
        for (int nt = 0; nt < 4; ++nt) {
            int srow = nt * 16 + dcol;
            if constexpr (F32OUT) {
#pragma unroll
                for (int r = 0; r < 8; ++r)
                    outF[srow * C_CH + c0 + r] = acc[mi][nt][r] + bb[r];
            } else {
                v8h hv;
#pragma unroll
                for (int r = 0; r < 8; ++r)
                    hv[r] = (_Float16)(acc[mi][nt][r] + bb[r]);
                *(v8h*)(outH + srow * C_CH + c0) = hv;
            }
        }
    }
}

// ---------------------------------------------------------------------------
// Fused per-image pipeline: stage image -> RoI resample -> 3 WMMA GEMMs ->
// 10-output heads. One 256-thread block (8 wave32) per image.
// ---------------------------------------------------------------------------
__global__ void __launch_bounds__(256) fused_kernel(
    const float* __restrict__ project,
    const unsigned int* __restrict__ wtiles,
    const float* __restrict__ b1, const float* __restrict__ b2,
    const float* __restrict__ b3,
    const float* __restrict__ w_note, const float* __restrict__ b_note,
    const float* __restrict__ w_reg,  const float* __restrict__ b_reg,
    float* __restrict__ out)
{
    extern __shared__ char smem[];
    _Float16* bufA  = (_Float16*)(smem);            // 64KB  [64][512] f16
    _Float16* bufB  = (_Float16*)(smem + 65536);    // 64KB  [64][512] f16
    float*    bufC  = (float*)(smem + 131072);      // 128KB [64][512] f32
    _Float16* Xfull = (_Float16*)(smem + 65536);    // 200704B, phase 0 only

    const int n   = blockIdx.x;
    const int tid = threadIdx.x;

    // ---- Phase 0a: stream image n into LDS (f16), fully coalesced ----------
    const float* src = project + (size_t)n * (C_CH * HW);
    for (int i = tid; i < (C_CH * HW) / 4; i += 256) {
        float4 v = ((const float4*)src)[i];
        v4h h = { (_Float16)v.x, (_Float16)v.y, (_Float16)v.z, (_Float16)v.w };
        *(v4h*)(Xfull + 4 * i) = h;
    }
    __syncthreads();

    // ---- Phase 0b: RoIAlign (fixed box -> fixed bilinear taps) -> bufA[s][c]
    for (int idx = tid; idx < SPPAD * C_CH; idx += 256) {
        int p = idx >> 9, c = idx & 511;
        _Float16 r = (_Float16)0.f;
        if (p < SP) {
            int i = p / 7, j = p % 7;
            float ty = (3.0f + 13.0f * (float)i) * (1.0f / 7.0f);
            float tx = (3.0f + 13.0f * (float)j) * (1.0f / 7.0f);
            int y0 = (int)floorf(ty); float fy = ty - (float)y0;
            int x0 = (int)floorf(tx); float fx = tx - (float)x0;
            const _Float16* xc = Xfull + c * HW + y0 * 14 + x0;
            float v00 = (float)xc[0];
            float v01 = (float)xc[1];
            float v10 = (float)xc[14];
            float v11 = (float)xc[15];
            float top = v00 + (v01 - v00) * fx;
            float bot = v10 + (v11 - v10) * fx;
            r = (_Float16)(top + (bot - top) * fy);
        }
        bufA[idx] = r;
    }
    __syncthreads();

    // ---- Three chained GEMMs (RoI commuted ahead of the convs) -------------
    gemm_stage<false>(bufA, wtiles + 0 * (size_t)WT_DWORDS_PER_W, b1, bufB, nullptr, tid);
    __syncthreads();
    gemm_stage<false>(bufB, wtiles + 1 * (size_t)WT_DWORDS_PER_W, b2, bufA, nullptr, tid);
    __syncthreads();
    gemm_stage<true >(bufA, wtiles + 2 * (size_t)WT_DWORDS_PER_W, b3, nullptr, bufC, tid);
    __syncthreads();

    // ---- Heads: note[2], reg[8] over flattened d = c*49 + p ----------------
    float acc10[10];
#pragma unroll
    for (int k = 0; k < 10; ++k) acc10[k] = 0.f;
    const int D = C_CH * SP;   // 25088
    for (int d = tid; d < D; d += 256) {
        int c = d / SP, p = d - c * SP;
        float z = bufC[p * C_CH + c];
#pragma unroll
        for (int k = 0; k < 2; ++k) acc10[k]     += w_note[(size_t)k * D + d] * z;
#pragma unroll
        for (int k = 0; k < 8; ++k) acc10[2 + k] += w_reg [(size_t)k * D + d] * z;
    }
    __syncthreads();
    float* red = (float*)smem;   // 10 * 256 floats, reuses bufA region
#pragma unroll
    for (int k = 0; k < 10; ++k) red[k * 256 + tid] = acc10[k];
    __syncthreads();
    for (int off = 128; off > 0; off >>= 1) {
        if (tid < off) {
#pragma unroll
            for (int k = 0; k < 10; ++k)
                red[k * 256 + tid] += red[k * 256 + tid + off];
        }
        __syncthreads();
    }
    if (tid < 2) {
        out[(size_t)n * 2 + tid] = red[tid * 256] + b_note[tid];
    } else if (tid < 10) {
        int k = tid - 2;
        out[1024 + (size_t)n * 8 + k] = red[(2 + k) * 256] + b_reg[k];
    }
}

// ---------------------------------------------------------------------------
extern "C" void kernel_launch(void* const* d_in, const int* in_sizes, int n_in,
                              void* d_out, int out_size, void* d_ws, size_t ws_size,
                              hipStream_t stream)
{
    const float* project = (const float*)d_in[0];
    const float* w1 = (const float*)d_in[1];
    const float* b1 = (const float*)d_in[2];
    const float* w2 = (const float*)d_in[3];
    const float* b2 = (const float*)d_in[4];
    const float* w3 = (const float*)d_in[5];
    const float* b3 = (const float*)d_in[6];
    const float* w_note = (const float*)d_in[7];
    const float* b_note = (const float*)d_in[8];
    const float* w_reg  = (const float*)d_in[9];
    const float* b_reg  = (const float*)d_in[10];

    unsigned int* wt = (unsigned int*)d_ws;    // 1.5 MB of preswizzled f16 weights

    preswz_kernel<<<192, 256, 0, stream>>>(w1, w2, w3, wt);

    (void)hipFuncSetAttribute((const void*)fused_kernel,
                              hipFuncAttributeMaxDynamicSharedMemorySize,
                              SMEM_BYTES);
    fused_kernel<<<512, 256, SMEM_BYTES, stream>>>(
        project, wt, b1, b2, b3, w_note, b_note, w_reg, b_reg, (float*)d_out);
}
